// myHMD_455266534064
// MI455X (gfx1250) — matrix operand order repaired and checked by reading, not compiled, not run on hardware
//
#include <hip/hip_runtime.h>
#include <hip/hip_bf16.h>

// ---------------------------------------------------------------------------
// Factored LSTM on MI455X (gfx1250, wave32, WMMA).
//
// Phase 1 (time-parallel): PreX = x @ [A_x;C_x]^T  (bf16 WMMA, f32 accum),
//                          SparX_g = rx_g @ B_x_g^T per gate.
// Phase 2 (recurrent, 512 steps): same two GEMM shapes with M=128 on h,
//                          then fused elementwise LSTM cell.
// GEMM staging uses double-buffered global_load_async_to_lds_b128 (ASYNCcnt)
// so the DMA of tile k+1 overlaps the WMMAs of tile k.
// ---------------------------------------------------------------------------

typedef __attribute__((ext_vector_type(16))) __bf16          v16bf;
typedef __attribute__((ext_vector_type(8)))  float           v8f;
typedef __attribute__((ext_vector_type(8)))  unsigned short  ushort8;
typedef __attribute__((ext_vector_type(4)))  int             v4i;

union BfFrag { v16bf v; ushort8 us[2]; };

#define AS1 __attribute__((address_space(1)))
#define AS3 __attribute__((address_space(3)))

#if __has_builtin(__builtin_amdgcn_global_load_async_to_lds_b128)
#define HAVE_ASYNC_LDS 1
#else
#define HAVE_ASYNC_LDS 0
#endif

__device__ __forceinline__ void wait_async0() {
#if __has_builtin(__builtin_amdgcn_s_wait_asynccnt)
    __builtin_amdgcn_s_wait_asynccnt(0);
#else
    asm volatile("s_wait_asynccnt 0x0" ::: "memory");
#endif
}

__device__ __forceinline__ unsigned short f2bf(float f) {
    unsigned int u = __float_as_uint(f);
    u += 0x7fffu + ((u >> 16) & 1u);   // round-to-nearest-even
    return (unsigned short)(u >> 16);
}

// ---------------------------------------------------------------------------
// Generic C[M,N](f32) = A[M,K](bf16, row-major) * W[N,K](bf16, row-major)^T
// Block tile 128x128x32, 256 threads = 8 waves, wave computes 64x32 (8 WMMA
// tiles of 16x16). Double-buffered async LDS staging. Optionally writes bf16
// copy of columns >= auxColStart. z-dim batches independent GEMMs.
// ntStore=1 -> streaming (non-temporal) f32 output, keeps L2 for hot data.
// Requires: 128|M, 128|N, 32|K (true for all shapes used here).
// ---------------------------------------------------------------------------
__global__ __launch_bounds__(256)
void gemm_bf16_nt(const unsigned short* __restrict__ A, int lda, long long strideAz,
                  const unsigned short* __restrict__ W, int ldw, long long strideWz,
                  float* __restrict__ C, int ldc, long long strideCz,
                  unsigned short* __restrict__ auxOut, int auxLd, int auxColStart,
                  long long strideAuxz, int K, int ntStore)
{
    const int z = blockIdx.z;
    A += (long long)z * strideAz;
    W += (long long)z * strideWz;
    C += (long long)z * strideCz;
    if (auxOut) auxOut += (long long)z * strideAuxz;

    const int n0 = blockIdx.x * 128;
    const int m0 = blockIdx.y * 128;

    // 32 bf16 per row + 8 pad (16B) keeps 16B-aligned rows, breaks bank strides
    __shared__ __align__(16) unsigned short As[2][128 * 40];
    __shared__ __align__(16) unsigned short Ws[2][128 * 40];

    const int tid  = threadIdx.x;
    const int wave = tid >> 5;
    const int lane = tid & 31;
    const int wm   = wave >> 2;   // 0..1 -> 64-row slab
    const int wn   = wave & 3;    // 0..3 -> 32-col slab

    const int lrow   = tid >> 1;        // 0..127 : staging row
    const int lchunk = (tid & 1) * 8;   // 0 or 8 : 16B chunk within row
    const int loff   = lrow * 40 + lchunk;

    const unsigned short* gA = &A[(long long)(m0 + lrow) * lda + lchunk];
    const unsigned short* gW = &W[(long long)(n0 + lrow) * ldw + lchunk];

    auto stage = [&](int buf, int k0) {
#if HAVE_ASYNC_LDS
        // Direct global -> LDS copy, tracked by ASYNCcnt (no VGPR round-trip)
        __builtin_amdgcn_global_load_async_to_lds_b128(
            (AS1 v4i*)(void*)(gA + k0), (AS3 v4i*)(void*)&As[buf][loff], 0, 0);
        __builtin_amdgcn_global_load_async_to_lds_b128(
            (AS1 v4i*)(void*)(gW + k0), (AS3 v4i*)(void*)&Ws[buf][loff], 0, 0);
#else
        *(ushort8*)&As[buf][loff] = *(const ushort8*)(gA + k0);
        *(ushort8*)&Ws[buf][loff] = *(const ushort8*)(gW + k0);
#endif
        if (k0 + 32 < K) {  // warm L2 for the tile after this one
            __builtin_prefetch(gA + k0 + 32, 0, 1);
            __builtin_prefetch(gW + k0 + 32, 0, 1);
        }
    };

    v8f acc[4][2] = {};

    stage(0, 0);                       // prologue: tile 0 in flight
    int cur = 0;
    for (int k0 = 0; k0 < K; k0 += 32, cur ^= 1) {
#if HAVE_ASYNC_LDS
        wait_async0();                 // my copies for buf[cur] have landed
#endif
        __syncthreads();               // everyone's landed; other buf free for reuse
        if (k0 + 32 < K) stage(cur ^ 1, k0 + 32);   // DMA overlaps compute below

        const int r = lane & 15;
        // A-matrix 16x32 bf16 lane layout (ISA 7.12.2):
        //   lanes 0-15: K {0..7, 16..23};  lanes 16-31: K {8..15, 24..31}
        const int ac0 = (lane < 16) ? 0 : 8;
        // B-matrix 32x16 lane layout: lanes 0-15: K {0..7, 16..23};  lanes 16-31: K {8..15, 24..31}
        const int bk0 = (lane < 16) ? 0 : 8;

        BfFrag a[4], b[2];
#pragma unroll
        for (int i = 0; i < 4; ++i) {
            const int row = (wm * 64 + i * 16 + r) * 40;
            a[i].us[0] = *(const ushort8*)&As[cur][row + ac0];
            a[i].us[1] = *(const ushort8*)&As[cur][row + ac0 + 16];
        }
#pragma unroll
        for (int j = 0; j < 2; ++j) {
            const int row = (wn * 32 + j * 16 + r) * 40;
            b[j].us[0] = *(const ushort8*)&Ws[cur][row + bk0];
            b[j].us[1] = *(const ushort8*)&Ws[cur][row + bk0 + 16];
        }
#pragma unroll
        for (int i = 0; i < 4; ++i)
#pragma unroll
            for (int j = 0; j < 2; ++j)
                acc[i][j] = __builtin_amdgcn_wmma_f32_16x16x32_bf16(
                    false, a[i].v, false, b[j].v, (short)0, acc[i][j], false, false);
    }

    // Epilogue: C/D 16x16 f32 layout (8 VGPRs): m = v + 8*(lane>=16), n = lane&15
    const int rl = lane & 15;
    const int hi = lane >> 4;
#pragma unroll
    for (int i = 0; i < 4; ++i)
#pragma unroll
        for (int j = 0; j < 2; ++j)
#pragma unroll
            for (int v = 0; v < 8; ++v) {
                const int m = m0 + wm * 64 + i * 16 + v + 8 * hi;
                const int n = n0 + wn * 32 + j * 16 + rl;
                const float val = acc[i][j][v];
                float* dst = &C[(long long)m * ldc + n];
                if (ntStore) __builtin_nontemporal_store(val, dst);
                else         *dst = val;
                if (auxOut && n >= auxColStart)
                    auxOut[(long long)m * auxLd + (n - auxColStart)] = f2bf(val);
            }
}

// ---------------------------------------------------------------------------
// Helpers
// ---------------------------------------------------------------------------
__global__ __launch_bounds__(256)
void convert_f32_bf16(unsigned short* __restrict__ dst, const float* __restrict__ src,
                      long long n)
{
    long long i = (long long)blockIdx.x * 256 + threadIdx.x;
    const long long stride = (long long)gridDim.x * 256;
    for (; i < n; i += stride) dst[i] = f2bf(src[i]);
}

__global__ __launch_bounds__(256)
void add_bias(float* __restrict__ bias, const float* __restrict__ bx,
              const float* __restrict__ bh)
{
    int i = blockIdx.x * 256 + threadIdx.x;
    if (i < 4096) bias[i] = bx[i] + bh[i];
}

__global__ __launch_bounds__(256)
void init_state(unsigned short* __restrict__ h_bf, float* __restrict__ c,
                const float* __restrict__ h0, const float* __restrict__ c0)
{
    int i = blockIdx.x * 256 + threadIdx.x;   // 128*1024 threads
    h_bf[i] = f2bf(h0[i]);
    c[i]    = c0[i];
}

// ---------------------------------------------------------------------------
// Fused LSTM cell: combine x-branch (precomputed, streamed NT) + h-branch +
// bias, gate nonlinearities, state update, emit h (f32 out + bf16 for GEMM).
// ---------------------------------------------------------------------------
__global__ __launch_bounds__(256)
void lstm_cell(const float* __restrict__ P1x_t,  // [B,2048] rich_x | rx_x
               const float* __restrict__ Sx_t,   // [B,3072] spar_x
               const float* __restrict__ R1h,    // [B,2048] rich_h | rx_h
               const float* __restrict__ Sh,     // [B,3072] spar_h
               const float* __restrict__ bias,   // [4096] gate-major
               float* __restrict__ c,            // [B,1024] state (in/out)
               float* __restrict__ out_t,        // d_out + t*B*H
               unsigned short* __restrict__ h_bf,
               float* __restrict__ h_final, float* __restrict__ c_final,
               int writeFinal)
{
    const int idx = blockIdx.x * 256 + threadIdx.x;  // B*H = 131072
    const int b = idx >> 10;
    const int h = idx & 1023;

    float pre[4];
#pragma unroll
    for (int g = 0; g < 4; ++g) {
        float p = bias[g * 1024 + h];
        if (h < 256) {
            const int o = b * 2048 + g * 256 + h;
            p += __builtin_nontemporal_load(&P1x_t[o]) + R1h[o];
        } else {
            const int o = b * 3072 + g * 768 + (h - 256);
            p += __builtin_nontemporal_load(&Sx_t[o]) + Sh[o];
        }
        pre[g] = p;
    }
    const float f  = 1.0f / (1.0f + __expf(-pre[0]));
    const float ii = 1.0f / (1.0f + __expf(-pre[1]));
    const float gg = tanhf(pre[2]);
    const float o  = 1.0f / (1.0f + __expf(-pre[3]));

    const float cn = f * c[idx] + ii * gg;
    const float hn = o * tanhf(cn);

    c[idx] = cn;
    __builtin_nontemporal_store(hn, &out_t[idx]);   // outs is a 256MB write-once stream
    h_bf[idx] = f2bf(hn);
    if (writeFinal) { h_final[idx] = hn; c_final[idx] = cn; }
}

// ---------------------------------------------------------------------------
// Host-side orchestration
// ---------------------------------------------------------------------------
extern "C" void kernel_launch(void* const* d_in, const int* in_sizes, int n_in,
                              void* d_out, int out_size, void* d_ws, size_t ws_size,
                              hipStream_t stream)
{
    (void)in_sizes; (void)n_in; (void)out_size; (void)ws_size;

    constexpr int T = 512, Bb = 128, H = 1024, IN = 1024, R = 256;
    constexpr long long TB = (long long)T * Bb;           // 65536
    constexpr int HR = H - R;                             // 768

    const float* x   = (const float*)d_in[0];
    const float* h0  = (const float*)d_in[1];
    const float* c0  = (const float*)d_in[2];
    const float* A_x = (const float*)d_in[3];
    const float* B_x = (const float*)d_in[4];
    const float* C_x = (const float*)d_in[5];
    const float* A_h = (const float*)d_in[6];
    const float* B_h = (const float*)d_in[7];
    const float* C_h = (const float*)d_in[8];
    const float* b_x = (const float*)d_in[9];
    const float* b_h = (const float*)d_in[10];

    float* out = (float*)d_out;
    float* outs    = out;                                   // [T,B,H]
    float* h_final = out + TB * H;                          // [B,H]
    float* c_final = h_final + (long long)Bb * H;           // [B,H]

    // -------- workspace carve --------
    char* p = (char*)d_ws;
    auto carve = [&](size_t bytes) -> void* {
        void* r = (void*)p;
        p += (bytes + 255) & ~(size_t)255;
        return r;
    };
    float*          bias = (float*)         carve(4096 * 4);
    unsigned short* Wx1  = (unsigned short*)carve((size_t)2048 * 1024 * 2); // [A_x;C_x]
    unsigned short* Bxw  = (unsigned short*)carve((size_t)3072 * 256 * 2);
    unsigned short* Wh1  = (unsigned short*)carve((size_t)2048 * 1024 * 2); // [A_h;C_h]
    unsigned short* Bhw  = (unsigned short*)carve((size_t)3072 * 256 * 2);
    unsigned short* h_bf = (unsigned short*)carve((size_t)Bb * H * 2);
    float*          cbuf = (float*)         carve((size_t)Bb * H * 4);
    float*          R1h  = (float*)         carve((size_t)Bb * 2048 * 4);
    unsigned short* RXh  = (unsigned short*)carve((size_t)Bb * 1024 * 2);
    float*          Sh   = (float*)         carve((size_t)Bb * 3072 * 4);
    unsigned short* Xbf  = (unsigned short*)carve((size_t)TB * IN * 2);
    unsigned short* RXx  = (unsigned short*)carve((size_t)TB * 1024 * 2);
    float*          P1x  = (float*)         carve((size_t)TB * 2048 * 4);
    float*          Sx   = (float*)         carve((size_t)TB * 3072 * 4);

    // -------- weight / input conversion to bf16 --------
    convert_f32_bf16<<<2048, 256, 0, stream>>>(Wx1,                A_x, (long long)4 * R * IN);
    convert_f32_bf16<<<2048, 256, 0, stream>>>(Wx1 + 1024 * 1024,  C_x, (long long)4 * R * IN);
    convert_f32_bf16<<<2048, 256, 0, stream>>>(Bxw,                B_x, (long long)4 * HR * R);
    convert_f32_bf16<<<2048, 256, 0, stream>>>(Wh1,                A_h, (long long)4 * R * H);
    convert_f32_bf16<<<2048, 256, 0, stream>>>(Wh1 + 1024 * 1024,  C_h, (long long)4 * R * H);
    convert_f32_bf16<<<2048, 256, 0, stream>>>(Bhw,                B_h, (long long)4 * HR * R);
    convert_f32_bf16<<<4096, 256, 0, stream>>>(Xbf,                x,   TB * IN);
    add_bias<<<16, 256, 0, stream>>>(bias, b_x, b_h);
    init_state<<<512, 256, 0, stream>>>(h_bf, cbuf, h0, c0);

    // -------- phase 1: time-parallel x-branch (streaming NT outputs) --------
    // P1x[TB,2048] = Xbf @ Wx1^T ; cols 1024.. re-emitted as bf16 into RXx
    gemm_bf16_nt<<<dim3(2048 / 128, TB / 128, 1), 256, 0, stream>>>(
        Xbf, IN, 0, Wx1, IN, 0, P1x, 2048, 0, RXx, 1024, 1024, 0, IN, 1);
    // Sx[TB, g*768..] = RXx[:, g*256..] @ B_x_g^T  (z = gate)
    gemm_bf16_nt<<<dim3(HR / 128, TB / 128, 4), 256, 0, stream>>>(
        RXx, 1024, 256, Bxw, 256, (long long)HR * 256, Sx, 3072, 768,
        nullptr, 0, 1 << 30, 0, R, 1);

    // -------- phase 2: recurrence (hot working set stays in L2) --------
    for (int t = 0; t < T; ++t) {
        gemm_bf16_nt<<<dim3(2048 / 128, 1, 1), 256, 0, stream>>>(
            h_bf, H, 0, Wh1, H, 0, R1h, 2048, 0, RXh, 1024, 1024, 0, H, 0);
        gemm_bf16_nt<<<dim3(HR / 128, 1, 4), 256, 0, stream>>>(
            RXh, 1024, 256, Bhw, 256, (long long)HR * 256, Sh, 3072, 768,
            nullptr, 0, 1 << 30, 0, R, 0);
        lstm_cell<<<512, 256, 0, stream>>>(
            P1x + (long long)t * Bb * 2048,
            Sx  + (long long)t * Bb * 3072,
            R1h, Sh, bias, cbuf,
            outs + (long long)t * Bb * H,
            h_bf, h_final, c_final, (t == T - 1) ? 1 : 0);
    }
}
